// LovaszHingeLoss_9242769621392
// MI455X (gfx1250) — matrix operand order repaired and checked by reading, not compiled, not run on hardware
//
#include <hip/hip_runtime.h>
#include <hip/hip_bf16.h>
#include <stdint.h>

typedef unsigned int  u32;
typedef unsigned char u8;
typedef __attribute__((ext_vector_type(16))) _Float16 v16h;
typedef __attribute__((ext_vector_type(8)))  float    v8f;

#define PIX     589824   // 768*768 pixels per image
#define NIMG    32
#define SORT_T  128      // threads per sort block
#define SORT_SEQ 8       // keys per thread (stable thread-serial counting)
#define SORT_BE 1024     // SORT_T*SORT_SEQ
#define SORT_NB 576      // PIX / SORT_BE
#define DOT_BE  4096     // elements per dot/scan block
#define DOT_NB  144      // PIX / DOT_BE

// ---- key encoding: ascending radix order == DESCENDING error; label in LSB ----
__device__ __forceinline__ u32 enc_key(float e, int lab) {
    u32 u  = __float_as_uint(e);
    u32 ka = (u & 0x80000000u) ? ~u : (u | 0x80000000u); // ascending uint == ascending float
    return ((~ka) & ~1u) | (u32)(lab & 1);
}
__device__ __forceinline__ float dec_err(u32 key) {
    u32 ka = ~key; // LSB carried the label: <=1 ulp perturbation of the error value
    u32 fb = (ka & 0x80000000u) ? (ka ^ 0x80000000u) : ~ka;
    return __uint_as_float(fb);
}

// ---------------- encode: errors -> sortable keys ----------------
__global__ void __launch_bounds__(256) k_encode(const float* __restrict__ x,
                                                const int* __restrict__ tg,
                                                u32* __restrict__ keys) {
    int gid = blockIdx.x * 256 + threadIdx.x;
    size_t base = (size_t)gid * 8;
#pragma unroll
    for (int j = 0; j < 8; ++j) {
        size_t i = base + j;
        float xv = x[i];
        int lab  = tg[i];
        float s  = lab ? 1.0f : -1.0f;
        float e  = 1.0f - xv * s;            // errors = 1 - logits*signs
        keys[i]  = enc_key(e, lab);
    }
}

// ---------------- radix pass 1: per-(image,block) 256-bin histogram ----------------
__global__ void __launch_bounds__(SORT_T) k_hist(const u32* __restrict__ src,
                                                 u32* __restrict__ hist, int shift) {
    __shared__ u32 h[256];
    int tid = threadIdx.x, blk = blockIdx.x, img = blockIdx.y;
    h[tid] = 0; h[tid + 128] = 0;
    __syncthreads();
    const u32* p = src + (size_t)img * PIX + (size_t)blk * SORT_BE + tid * SORT_SEQ;
    __builtin_prefetch(p, 0, 1);             // global_prefetch_b8
#pragma unroll
    for (int j = 0; j < SORT_SEQ; ++j) atomicAdd(&h[(p[j] >> shift) & 255u], 1u);
    __syncthreads();
    hist[((size_t)img * 256 + tid) * SORT_NB + blk] = h[tid];
    hist[((size_t)img * 256 + tid + 128) * SORT_NB + blk] = h[tid + 128];
}

// ---------------- radix pass 2: per-image digit bases + per-block offsets (in place) ----------------
__global__ void __launch_bounds__(256) k_digit_scan(u32* __restrict__ hist) {
    __shared__ u32 tot[256];
    int t = threadIdx.x, img = blockIdx.x;
    u32* hp = hist + ((size_t)img * 256 + t) * SORT_NB;
    u32 s = 0;
    for (int b = 0; b < SORT_NB; ++b) s += hp[b];
    tot[t] = s;
    __syncthreads();
    for (int off = 1; off < 256; off <<= 1) {        // Hillis-Steele inclusive scan over bins
        u32 v = (t >= off) ? tot[t - off] : 0u;
        __syncthreads();
        tot[t] += v;
        __syncthreads();
    }
    u32 run = tot[t] - s;                            // exclusive digit base
    for (int b = 0; b < SORT_NB; ++b) { u32 o = hp[b]; hp[b] = run; run += o; }
}

// ---------------- radix pass 3: STABLE scatter (thread-serial counting in LDS) ----------------
__global__ void __launch_bounds__(SORT_T) k_scatter(const u32* __restrict__ src,
                                                    u32* __restrict__ dst,
                                                    const u32* __restrict__ hist, int shift) {
    __shared__ u8  cnt[256][SORT_T];   // 32 KB: per-(digit,thread) counts / intra-wave prefixes
    __shared__ u32 wbase[256][4];      //  4 KB: digit global base + wave-exclusive prefix
    int tid = threadIdx.x, blk = blockIdx.x, img = blockIdx.y;

    u32* cz = (u32*)&cnt[0][0];
    for (int i = tid; i < 256 * SORT_T / 4; i += SORT_T) cz[i] = 0;
    __syncthreads();

    u32 myk[SORT_SEQ];
    const u32* p = src + (size_t)img * PIX + (size_t)blk * SORT_BE + tid * SORT_SEQ;
#pragma unroll
    for (int j = 0; j < SORT_SEQ; ++j) { myk[j] = p[j]; cnt[(myk[j] >> shift) & 255u][tid]++; }
    __syncthreads();

    // each thread owns two digit rows: intra-wave exclusive prefixes (<=248, fits u8) + wave bases
    for (int bb = 0; bb < 2; ++bb) {
        int b = tid + bb * 128;
        u32 bofs = hist[((size_t)img * 256 + b) * SORT_NB + blk];
        u32 wacc = 0;
        for (int w = 0; w < 4; ++w) {
            u32 wsum = 0;
            for (int tt = w * 32; tt < w * 32 + 32; ++tt) {
                u32 c = cnt[b][tt]; cnt[b][tt] = (u8)wsum; wsum += c;
            }
            wbase[b][w] = bofs + wacc;
            wacc += wsum;
        }
    }
    __syncthreads();

    u32* out = dst + (size_t)img * PIX;
#pragma unroll
    for (int j = 0; j < SORT_SEQ; ++j) {
        u32 k = myk[j], d = (k >> shift) & 255u;
        u32 r = cnt[d][tid]; cnt[d][tid] = (u8)(r + 1);   // thread-private column: stable
        out[wbase[d][tid >> 5] + r] = k;
    }
}

// ---------------- block sums of sorted labels ----------------
__global__ void __launch_bounds__(256) k_bsum(const u32* __restrict__ keys,
                                              float* __restrict__ bsum) {
    __shared__ float red[256];
    int tid = threadIdx.x, blk = blockIdx.x, img = blockIdx.y;
    const u32* p = keys + (size_t)img * PIX + (size_t)blk * DOT_BE + tid * 16;
    u32 s = 0;
#pragma unroll
    for (int j = 0; j < 16; ++j) s += p[j] & 1u;
    red[tid] = (float)s;
    __syncthreads();
    for (int o = 128; o > 0; o >>= 1) { if (tid < o) red[tid] += red[tid + o]; __syncthreads(); }
    if (tid == 0) bsum[img * DOT_NB + blk] = red[0];
}

// ---------------- per-image exclusive scan of block sums, and gts totals ----------------
__global__ void __launch_bounds__(256) k_bscan(const float* __restrict__ bsum,
                                               float* __restrict__ boff,
                                               float* __restrict__ gts) {
    __shared__ float sc[256];
    int t = threadIdx.x, img = blockIdx.x;
    float v = (t < DOT_NB) ? bsum[img * DOT_NB + t] : 0.f;
    sc[t] = v;
    __syncthreads();
    for (int off = 1; off < 256; off <<= 1) {
        float u = (t >= off) ? sc[t - off] : 0.f;
        __syncthreads();
        sc[t] += u;
        __syncthreads();
    }
    if (t < DOT_NB) boff[img * DOT_NB + t] = sc[t] - v;   // exclusive
    if (t == DOT_NB - 1) gts[img] = sc[t];                // total positives
}

// ---------------- WMMA segmented scan + Lovasz gradient dot ----------------
// cumsum over each 16-run = L(16x16 lower-tri ones) x G  via V_WMMA_F32_16X16X32_F16.
__global__ void __launch_bounds__(256) k_dot(const u32* __restrict__ keys,
                                             const float* __restrict__ boff,
                                             const float* __restrict__ gts,
                                             float* __restrict__ perimg) {
    __shared__ float colsum[256];
    __shared__ float red[256];
    int tid = threadIdx.x, blk = blockIdx.x, img = blockIdx.y;
    int w = tid >> 5, lane = tid & 31;        // wave32
    int m = lane & 15, half = lane >> 4;

    // A fragment = [L | 0]: 16x32 f16. Layout: slot s -> K = half*16+s; half 0 holds K<16, half 1 holds K>=16 (zero).
    v16h a;
#pragma unroll
    for (int s = 0; s < 16; ++s) {
        float av = 0.f;
        if (half == 0 && s <= m) av = 1.f;
        a[s] = (_Float16)av;
    }

    const u32* kb = keys + (size_t)img * PIX + (size_t)blk * DOT_BE + w * 512;
    __builtin_prefetch(kb + lane * 16, 0, 1);

    v8f d0, d1;
#pragma unroll
    for (int tile = 0; tile < 2; ++tile) {
        // B fragment 32x16 f16: lanes 0..15 hold column n=lane, K=0..15 (consecutive labels); rest zero.
        v16h b;
#pragma unroll
        for (int s = 0; s < 16; ++s) b[s] = (_Float16)0.f;
        if (lane < 16) {
            const uint4* kp = (const uint4*)(kb + tile * 256 + lane * 16);
            u32 q[16];
#pragma unroll
            for (int v4 = 0; v4 < 4; ++v4) {
                uint4 qq = kp[v4];
                q[v4 * 4] = qq.x; q[v4 * 4 + 1] = qq.y; q[v4 * 4 + 2] = qq.z; q[v4 * 4 + 3] = qq.w;
            }
#pragma unroll
            for (int s = 0; s < 16; ++s) b[s] = (_Float16)(float)(q[s] & 1u);
        }
        v8f c;
#pragma unroll
        for (int i = 0; i < 8; ++i) c[i] = 0.f;
        // EXEC is all-ones here (no divergence at call site), as WMMA requires.
        v8f d = __builtin_amdgcn_wmma_f32_16x16x32_f16(false, a, false, b, (short)0, c,
                                                       false, false);
        if (tile == 0) d0 = d; else d1 = d;
        // D layout: M=15 row lives in VGPR7 of lanes 16..31 (N = lane-16) -> column sums
        if (lane >= 16) colsum[w * 32 + tile * 16 + (lane - 16)] = d[7];
    }
    __syncthreads();
    if (tid == 0) {                                  // exclusive scan of 256 column sums
        float run = 0.f;
        for (int i = 0; i < 256; ++i) { float v = colsum[i]; colsum[i] = run; run += v; }
    }
    __syncthreads();

    float G  = gts[img];
    float bo = boff[img * DOT_NB + blk];
    int n  = lane & 15;
    int mb = (lane < 16) ? 0 : 8;                    // D layout: lane<16 holds M=r, lane>=16 holds M=r+8
    float acc = 0.f;
#pragma unroll
    for (int tile = 0; tile < 2; ++tile) {
        v8f d = (tile == 0) ? d0 : d1;
        float colEx  = colsum[w * 32 + tile * 16 + n] + bo;
        int   erank0 = blk * DOT_BE + w * 512 + tile * 256 + n * 16 + mb;
        const uint4* kp = (const uint4*)(kb + tile * 256 + n * 16 + mb);
        uint4 qa = kp[0], qb = kp[1];
        u32 q[8] = {qa.x, qa.y, qa.z, qa.w, qb.x, qb.y, qb.z, qb.w};
#pragma unroll
        for (int r = 0; r < 8; ++r) {
            u32  key = q[r];
            float g  = (float)(key & 1u);
            float err = dec_err(key);
            float cp  = d[r] + colEx;                // inclusive cumsum of gt at rank i
            int   i   = erank0 + r;
            float nn1 = (float)(i + 1);
            float jac1 = 1.f - (G - cp) / (G + nn1 - cp);
            float prev = 0.f;
            if (i > 0) { float cpp = cp - g; prev = 1.f - (G - cpp) / (G + (float)i - cpp); }
            float re = err > 0.f ? err : 0.f;        // relu(errors_sorted)
            acc += re * (jac1 - prev);               // grad_i = jac_i - jac_{i-1}
        }
    }
    red[tid] = acc;
    __syncthreads();
    for (int o = 128; o > 0; o >>= 1) { if (tid < o) red[tid] += red[tid + o]; __syncthreads(); }
    if (tid == 0) atomicAdd(&perimg[img], red[0]);
}

__global__ void k_init(float* perimg) { if (threadIdx.x < NIMG) perimg[threadIdx.x] = 0.f; }

__global__ void k_final(const float* __restrict__ perimg, const float* __restrict__ gts,
                        float* __restrict__ out) {
    if (threadIdx.x == 0) {
        float s = 0.f;
        for (int i = 0; i < NIMG; ++i) s += (gts[i] > 0.f) ? perimg[i] : 0.f; // empty-mask weight
        out[0] = s / (float)NIMG;
    }
}

extern "C" void kernel_launch(void* const* d_in, const int* in_sizes, int n_in,
                              void* d_out, int out_size, void* d_ws, size_t ws_size,
                              hipStream_t stream) {
    const float* x  = (const float*)d_in[0];  // [32,1,768,768] f32 logits
    const int*   tg = (const int*)d_in[1];    // [32,1,768,768] i32 labels
    float* out = (float*)d_out;               // scalar f32
    char* ws = (char*)d_ws;

    // workspace layout (~170 MB; key set is L2-resident on MI455X's 192 MB L2)
    u32*   keysA  = (u32*)(ws);
    u32*   keysB  = (u32*)(ws + 75497472ull);                    // 32*PIX*4
    u32*   hist   = (u32*)(ws + 150994944ull);                   // 32*256*576*4 = 18.9 MB
    float* bsum   = (float*)(ws + 169869312ull);
    float* boff   = (float*)(ws + 169887744ull);
    float* gts    = (float*)(ws + 169906176ull);
    float* perimg = (float*)(ws + 169906432ull);

    k_init<<<1, 32, 0, stream>>>(perimg);
    k_encode<<<dim3(9216), dim3(256), 0, stream>>>(x, tg, keysA);

    u32* src = keysA; u32* dst = keysB;
    for (int p = 0; p < 4; ++p) {                 // LSD radix sort, 8-bit digits, per image
        int shift = p * 8;
        k_hist<<<dim3(SORT_NB, NIMG), dim3(SORT_T), 0, stream>>>(src, hist, shift);
        k_digit_scan<<<dim3(NIMG), dim3(256), 0, stream>>>(hist);
        k_scatter<<<dim3(SORT_NB, NIMG), dim3(SORT_T), 0, stream>>>(src, dst, hist, shift);
        u32* t = src; src = dst; dst = t;
    }
    // 4 passes -> sorted keys back in keysA (== src)
    k_bsum<<<dim3(DOT_NB, NIMG), dim3(256), 0, stream>>>(src, bsum);
    k_bscan<<<dim3(NIMG), dim3(256), 0, stream>>>(bsum, boff, gts);
    k_dot<<<dim3(DOT_NB, NIMG), dim3(256), 0, stream>>>(src, boff, gts, perimg);
    k_final<<<1, 32, 0, stream>>>(perimg, gts, out);
}
